// MultiHeadAttention_40329742910041
// MI455X (gfx1250) — matrix operand order repaired and checked by reading, not compiled, not run on hardware
//
#include <hip/hip_runtime.h>

typedef __attribute__((ext_vector_type(16))) __bf16 v16bf;
typedef __attribute__((ext_vector_type(8)))  float  v8f;
typedef int v4i_vs __attribute__((vector_size(16)));   // matches builtin's int4 pointee

#define HEADS  16
#define HD     128
#define SEQ    1024
#define BATCH  4
#define DMODEL 2048
#define QTILE  128   // query rows per block
#define WAVES  8
#define KTILE  32    // key rows per iteration

union Frag { v16bf v; uint4 q[2]; };

static __device__ __forceinline__ unsigned short f2bf(float f) {
  unsigned int u = __builtin_bit_cast(unsigned int, f);
  u += 0x7FFFu + ((u >> 16) & 1u);               // round-to-nearest-even
  return (unsigned short)(u >> 16);
}
static __device__ __forceinline__ __bf16 f2bfh(float f) {
  unsigned short h = f2bf(f);
  return __builtin_bit_cast(__bf16, h);
}

// ---- CDNA5 async global->LDS copy (ASYNCcnt path) ----
static __device__ __forceinline__ void async_cp_b128(unsigned short* dst,
                                                     const unsigned short* src) {
#if __has_builtin(__builtin_amdgcn_global_load_async_to_lds_b128)
  __builtin_amdgcn_global_load_async_to_lds_b128(
      (__attribute__((address_space(1))) v4i_vs*)src,
      (__attribute__((address_space(3))) v4i_vs*)dst, 0, 0);
#else
  unsigned int ldsoff =
      (unsigned int)(size_t)((__attribute__((address_space(3))) unsigned short*)dst);
  asm volatile("global_load_async_to_lds_b128 %0, %1, off"
               :: "v"(ldsoff), "v"(src) : "memory");
#endif
}
static __device__ __forceinline__ void wait_asynccnt0() {
#if __has_builtin(__builtin_amdgcn_s_wait_asynccnt)
  __builtin_amdgcn_s_wait_asynccnt(0);
#else
  asm volatile("s_wait_asynccnt 0x0" ::: "memory");
#endif
}

// ---- pre-pass: fp32 K/V -> bf16, head-major K and transposed V, into d_ws ----
__global__ __launch_bounds__(256)
void kv_convert_kernel(const float* __restrict__ K, const float* __restrict__ V,
                       unsigned short* __restrict__ K16, unsigned short* __restrict__ VT16) {
  const unsigned int gid = blockIdx.x * 256u + threadIdx.x; // 2^20 threads
  const int d8 = gid & 15;
  const int h  = (gid >> 4) & 15;
  const int s  = (gid >> 8) & 1023;
  const int b  = gid >> 18;
  const size_t src = ((size_t)(b * SEQ + s)) * DMODEL + (size_t)h * HD + d8 * 8;
  float buf[8];
  *(float4*)&buf[0] = *(const float4*)(K + src);
  *(float4*)&buf[4] = *(const float4*)(K + src + 4);
  uint4 pk;
  pk.x = (unsigned int)f2bf(buf[0]) | ((unsigned int)f2bf(buf[1]) << 16);
  pk.y = (unsigned int)f2bf(buf[2]) | ((unsigned int)f2bf(buf[3]) << 16);
  pk.z = (unsigned int)f2bf(buf[4]) | ((unsigned int)f2bf(buf[5]) << 16);
  pk.w = (unsigned int)f2bf(buf[6]) | ((unsigned int)f2bf(buf[7]) << 16);
  *(uint4*)&K16[(((size_t)(b * HEADS + h)) * SEQ + s) * HD + d8 * 8] = pk;

  *(float4*)&buf[0] = *(const float4*)(V + src);
  *(float4*)&buf[4] = *(const float4*)(V + src + 4);
  const size_t vtb = ((size_t)(b * HEADS + h)) * HD * SEQ;
#pragma unroll
  for (int c = 0; c < 8; ++c)
    VT16[vtb + (size_t)(d8 * 8 + c) * SEQ + s] = f2bf(buf[c]);
}

// ---- flash-attention forward, causal, bf16 WMMA ----
template <bool USE_ASYNC>
__global__ __launch_bounds__(256)
void fa_fwd_causal_kernel(const float* __restrict__ Q,
                          const float* __restrict__ Kf, const float* __restrict__ Vf,
                          const unsigned short* __restrict__ Kbf,
                          const unsigned short* __restrict__ Vtbf,
                          float* __restrict__ O) {
  __shared__ __align__(16) unsigned short Klds[KTILE * HD];         // [key][hd] bf16
  __shared__ __align__(16) unsigned short VTlds[HD * KTILE];        // [hd][key] bf16
  __shared__ __align__(16) unsigned short Plds[WAVES][16 * KTILE];  // per-wave P 16x32 bf16

  const int tid  = threadIdx.x;
  const int lane = tid & 31;
  const int w    = tid >> 5;
  const int qc   = blockIdx.x;      // q chunk
  const int h    = blockIdx.y;      // head
  const int b    = blockIdx.z;      // batch

  const int qb0   = qc * QTILE;
  const int qbase = qb0 + w * 16;   // this wave's first query row
  const int n     = lane & 15;
  const int half  = lane >> 4;
  const int mbase = half * 8;       // C layout row base
  const int kb16  = half * 16;      // unused; B loads use kb8 too
  const int kb8   = half * 8;       // A layout K base

  const float scale = 0.08838834764831845f;  // 1/sqrt(128)

  // ---- Q tile -> 4 A-fragments (16x32 bf16), scale folded in ----
  const float* qrow = Q + ((size_t)(b * SEQ + qbase + n)) * DMODEL + (size_t)h * HD;
  v16bf qa[4];
#pragma unroll
  for (int kc = 0; kc < 4; ++kc) {
    float buf[16];
    const float* p0 = qrow + kc * 32 + kb8;
    const float* p1 = p0 + 16;
    *(float4*)&buf[0]  = *(const float4*)(p0);
    *(float4*)&buf[4]  = *(const float4*)(p0 + 4);
    *(float4*)&buf[8]  = *(const float4*)(p1);
    *(float4*)&buf[12] = *(const float4*)(p1 + 4);
    v16bf qv;
#pragma unroll
    for (int i = 0; i < 16; ++i) qv[i] = f2bfh(buf[i] * scale);
    qa[kc] = qv;
  }

  float mstat[8], lstat[8];
  v8f zero = {};
  v8f acc[8];
#pragma unroll
  for (int r = 0; r < 8; ++r) { mstat[r] = -1e30f; lstat[r] = 0.0f; }
#pragma unroll
  for (int n2 = 0; n2 < 8; ++n2) acc[n2] = zero;

  const int ntiles = qb0 / KTILE + QTILE / KTILE;
  const int wmax_t = (qbase + 15) / KTILE;

  for (int t = 0; t < ntiles; ++t) {
    const int j = t * KTILE;
    __syncthreads();
    if constexpr (USE_ASYNC) {
      // ---- stage pre-converted bf16 tiles with CDNA5 async LDS copies ----
      const unsigned short* ksrc = Kbf + (((size_t)(b * HEADS + h)) * SEQ + j) * HD;
      const unsigned short* vsrc = Vtbf + ((size_t)(b * HEADS + h)) * HD * SEQ + j;
#pragma unroll
      for (int i = 0; i < 2; ++i) {
        const int c = tid + i * 256;          // 16B chunk id, 0..511
        async_cp_b128(&Klds[c * 8], ksrc + c * 8);
        const int row = c >> 2, part = c & 3; // VT: 128 rows x 64B
        async_cp_b128(&VTlds[row * KTILE + part * 8],
                      vsrc + (size_t)row * SEQ + part * 8);
      }
      wait_asynccnt0();
    } else {
      // ---- fallback: load fp32 and convert in-kernel ----
#pragma unroll
      for (int it = 0; it < 4; ++it) {
        const int fid = tid + it * 256;
        const int row = fid >> 5;
        const int c4  = fid & 31;
        const size_t goff = ((size_t)(b * SEQ + j + row)) * DMODEL + (size_t)h * HD + c4 * 4;
        float4 kv = *(const float4*)(Kf + goff);
        uint2 pk;
        pk.x = (unsigned int)f2bf(kv.x) | ((unsigned int)f2bf(kv.y) << 16);
        pk.y = (unsigned int)f2bf(kv.z) | ((unsigned int)f2bf(kv.w) << 16);
        *(uint2*)&Klds[row * HD + c4 * 4] = pk;
        float4 vv = *(const float4*)(Vf + goff);
        VTlds[(c4 * 4 + 0) * KTILE + row] = f2bf(vv.x);
        VTlds[(c4 * 4 + 1) * KTILE + row] = f2bf(vv.y);
        VTlds[(c4 * 4 + 2) * KTILE + row] = f2bf(vv.z);
        VTlds[(c4 * 4 + 3) * KTILE + row] = f2bf(vv.w);
      }
    }
    __syncthreads();

    if (t <= wmax_t) {
      // ---- scores S = (Q*scale) . K^T -> two 16x16 f32 tiles ----
      v8f s0 = zero, s1 = zero;
#pragma unroll
      for (int kc = 0; kc < 4; ++kc) {
        const int ko = kc * 32 + kb8;
        Frag fb0, fb1;
        fb0.q[0] = *(const uint4*)&Klds[(0  + n) * HD + ko];
        fb0.q[1] = *(const uint4*)&Klds[(0  + n) * HD + ko + 16];
        fb1.q[0] = *(const uint4*)&Klds[(16 + n) * HD + ko];
        fb1.q[1] = *(const uint4*)&Klds[(16 + n) * HD + ko + 16];
        s0 = __builtin_amdgcn_wmma_f32_16x16x32_bf16(false, qa[kc], false, fb0.v,
                                                     (short)0, s0, false, false);
        s1 = __builtin_amdgcn_wmma_f32_16x16x32_bf16(false, qa[kc], false, fb1.v,
                                                     (short)0, s1, false, false);
      }
      // ---- causal mask (key <= query) ----
#pragma unroll
      for (int r = 0; r < 8; ++r) {
        const int row = qbase + mbase + r;
        if (j + n      > row) s0[r] = -1e30f;
        if (j + 16 + n > row) s1[r] = -1e30f;
      }
      // ---- online softmax ----
      float alpha[8];
#pragma unroll
      for (int r = 0; r < 8; ++r) {
        float mx = fmaxf(s0[r], s1[r]);
        mx = fmaxf(mx, __shfl_xor(mx, 1));
        mx = fmaxf(mx, __shfl_xor(mx, 2));
        mx = fmaxf(mx, __shfl_xor(mx, 4));
        mx = fmaxf(mx, __shfl_xor(mx, 8));
        const float mnew = fmaxf(mstat[r], mx);
        alpha[r] = __expf(mstat[r] - mnew);
        const float p0 = __expf(s0[r] - mnew);
        const float p1 = __expf(s1[r] - mnew);
        s0[r] = p0; s1[r] = p1;
        float rs = p0 + p1;
        rs += __shfl_xor(rs, 1);
        rs += __shfl_xor(rs, 2);
        rs += __shfl_xor(rs, 4);
        rs += __shfl_xor(rs, 8);
        lstat[r] = lstat[r] * alpha[r] + rs;
        mstat[r] = mnew;
      }
#pragma unroll
      for (int n2 = 0; n2 < 8; ++n2)
#pragma unroll
        for (int r = 0; r < 8; ++r) acc[n2][r] *= alpha[r];

      // ---- P (C layout) -> LDS row-major -> A fragment ----
      unsigned short* pw = &Plds[w][0];
#pragma unroll
      for (int r = 0; r < 8; ++r) {
        pw[(mbase + r) * KTILE + n]      = f2bf(s0[r]);
        pw[(mbase + r) * KTILE + 16 + n] = f2bf(s1[r]);
      }
      asm volatile("s_wait_dscnt 0" ::: "memory");
      Frag pa;
      pa.q[0] = *(const uint4*)&pw[n * KTILE + kb8];
      pa.q[1] = *(const uint4*)&pw[n * KTILE + 16 + kb8];

      // ---- O += P . V ----
#pragma unroll
      for (int n2 = 0; n2 < 8; ++n2) {
        Frag vb;
        vb.q[0] = *(const uint4*)&VTlds[(n2 * 16 + n) * KTILE + kb8];
        vb.q[1] = *(const uint4*)&VTlds[(n2 * 16 + n) * KTILE + 16 + kb8];
        acc[n2] = __builtin_amdgcn_wmma_f32_16x16x32_bf16(false, pa.v, false, vb.v,
                                                          (short)0, acc[n2], false, false);
      }
    }
  }

  // ---- epilogue ----
  float rinv[8];
#pragma unroll
  for (int r = 0; r < 8; ++r) rinv[r] = 1.0f / lstat[r];
#pragma unroll
  for (int n2 = 0; n2 < 8; ++n2) {
#pragma unroll
    for (int r = 0; r < 8; ++r) {
      const size_t idx = ((size_t)(b * SEQ + qbase + mbase + r)) * DMODEL
                       + (size_t)h * HD + n2 * 16 + n;
      O[idx] = acc[n2][r] * rinv[r];
    }
  }
}

extern "C" void kernel_launch(void* const* d_in, const int* in_sizes, int n_in,
                              void* d_out, int out_size, void* d_ws, size_t ws_size,
                              hipStream_t stream) {
  (void)in_sizes; (void)n_in; (void)out_size;
  const float* q = (const float*)d_in[0];
  const float* k = (const float*)d_in[1];
  const float* v = (const float*)d_in[2];
  float* o = (float*)d_out;
  dim3 grid(SEQ / QTILE, HEADS, BATCH);

  const size_t kvHalf = (size_t)BATCH * HEADS * SEQ * HD;      // elements per tensor
  if (ws_size >= 2 * kvHalf * sizeof(unsigned short)) {
    unsigned short* k16  = (unsigned short*)d_ws;
    unsigned short* vt16 = k16 + kvHalf;
    kv_convert_kernel<<<dim3(4096), dim3(256), 0, stream>>>(k, v, k16, vt16);
    fa_fwd_causal_kernel<true><<<grid, dim3(256), 0, stream>>>(q, nullptr, nullptr,
                                                               k16, vt16, o);
  } else {
    fa_fwd_causal_kernel<false><<<grid, dim3(256), 0, stream>>>(q, k, v,
                                                                nullptr, nullptr, o);
  }
}
